// LGemAttention_16234976379545
// MI455X (gfx1250) — matrix operand order repaired and checked by reading, not compiled, not run on hardware
//
#include <hip/hip_runtime.h>

#define B_  2
#define T_  2048
#define C_  2048
#define H_  16
#define D_  128
#define NT_ (B_*T_)   // 4096

typedef __attribute__((ext_vector_type(16))) __bf16 bf16x16;
typedef __attribute__((ext_vector_type(8)))  float  floatx8;

union Frag { bf16x16 v; unsigned int u[8]; };

#if __has_builtin(__builtin_amdgcn_global_load_async_to_lds_b128)
#define HAVE_ASYNC_LDS 1
typedef int v4i __attribute__((vector_size(16)));
typedef __attribute__((address_space(1))) v4i* gv4i_p;   // global int4*
typedef __attribute__((address_space(3))) v4i* lv4i_p;   // LDS int4*
#endif

__device__ __forceinline__ unsigned short f32_to_bf16(float f) {
  unsigned int u = __float_as_uint(f);
  u += 0x7FFFu + ((u >> 16) & 1u);          // round-to-nearest-even
  return (unsigned short)(u >> 16);
}
__device__ __forceinline__ float bf16_to_f32(unsigned short h) {
  return __uint_as_float(((unsigned int)h) << 16);
}
__device__ __forceinline__ floatx8 zero8() {
  floatx8 z;
#pragma unroll
  for (int i = 0; i < 8; ++i) z[i] = 0.f;
  return z;
}

// 16-byte memory -> LDS copy; async (ASYNCcnt) when the builtin exists.
__device__ __forceinline__ void copy16_to_lds(unsigned short* lds_dst,
                                              const unsigned short* gsrc) {
#ifdef HAVE_ASYNC_LDS
  __builtin_amdgcn_global_load_async_to_lds_b128(
      (gv4i_p)gsrc, (lv4i_p)lds_dst, 0, 0);
#else
  *(uint4*)lds_dst = *(const uint4*)gsrc;
#endif
}
__device__ __forceinline__ void wait_async_lds() {
#ifdef HAVE_ASYNC_LDS
  asm volatile("s_wait_asynccnt 0x0" ::: "memory");
#endif
}

// ---------------------------------------------------------------------------
// fp32 -> bf16 conversion, 4 elements per thread
// ---------------------------------------------------------------------------
__global__ void cvt_f32_to_bf16_kernel(const float* __restrict__ src,
                                       unsigned short* __restrict__ dst, int n4) {
  const int i = blockIdx.x * blockDim.x + threadIdx.x;
  if (i < n4) {
    const float4 f = ((const float4*)src)[i];
    __align__(8) unsigned short o[4];
    o[0] = f32_to_bf16(f.x); o[1] = f32_to_bf16(f.y);
    o[2] = f32_to_bf16(f.z); o[3] = f32_to_bf16(f.w);
    ((uint2*)dst)[i] = *reinterpret_cast<uint2*>(o);
  }
}

// ---------------------------------------------------------------------------
// Generic bf16 GEMM: D(MxN) = A(MxK) * B(KxN), 128x128 tile, K-step 64.
// 256 threads = 8 waves (4m x 2n); each wave: 32x64 via 2x4 WMMA tiles.
// A tile staged via async memory->LDS; B tile transposed to [n][k] in LDS.
// ---------------------------------------------------------------------------
template <int OUT_F32>
__global__ __launch_bounds__(256) void gemm_bf16_128x128(
    const unsigned short* __restrict__ A,
    const unsigned short* __restrict__ Bm,
    void* __restrict__ Dst,
    int M, int N, int K)
{
  __shared__ unsigned short As[128][64];   // [m][k]   16 KiB
  __shared__ unsigned short Bs[128][64];   // [n][k]   16 KiB (transposed stage)

  const int tid  = threadIdx.x;
  const int lane = tid & 31;
  const int wid  = tid >> 5;
  const int wm   = wid >> 1, wn = wid & 1;
  const int g    = lane >> 4, hl = lane & 15;

  const int m0 = blockIdx.y * 128;
  const int n0 = blockIdx.x * 128;

  floatx8 acc[2][4];
#pragma unroll
  for (int i = 0; i < 2; ++i)
#pragma unroll
    for (int j = 0; j < 4; ++j) acc[i][j] = zero8();

  const int ar = tid >> 2;          // 0..63  (rows ar, ar+64)
  const int ac = (tid & 3) * 16;    // 16 contiguous k
  const int bk = tid >> 2;          // 0..63  (k row of B)
  const int bn = (tid & 3) * 32;    // 32 contiguous n

  for (int k0 = 0; k0 < K; k0 += 64) {
    __syncthreads();
    // ---- A tile: 4 async B128 transfers per thread ----
    copy16_to_lds(&As[ar][ac],          A + (size_t)(m0 + ar) * K + k0 + ac);
    copy16_to_lds(&As[ar][ac + 8],      A + (size_t)(m0 + ar) * K + k0 + ac + 8);
    copy16_to_lds(&As[ar + 64][ac],     A + (size_t)(m0 + ar + 64) * K + k0 + ac);
    copy16_to_lds(&As[ar + 64][ac + 8], A + (size_t)(m0 + ar + 64) * K + k0 + ac + 8);
    // ---- B tile: load 32 contiguous n, scatter transposed ----
    {
      __align__(16) unsigned short tmp[32];
      const unsigned short* src = Bm + (size_t)(k0 + bk) * N + n0 + bn;
      *(uint4*)&tmp[0]  = *(const uint4*)(src);
      *(uint4*)&tmp[8]  = *(const uint4*)(src + 8);
      *(uint4*)&tmp[16] = *(const uint4*)(src + 16);
      *(uint4*)&tmp[24] = *(const uint4*)(src + 24);
#pragma unroll
      for (int i = 0; i < 32; ++i) Bs[bn + i][bk] = tmp[i];
    }
    wait_async_lds();
    __syncthreads();

#pragma unroll
    for (int ks = 0; ks < 2; ++ks) {
      Frag a[2], bf[4];
#pragma unroll
      for (int mf = 0; mf < 2; ++mf) {
        const int row = wm * 32 + mf * 16 + hl;
#pragma unroll
        for (int v = 0; v < 8; ++v)
          a[mf].u[v] = *(const unsigned int*)
              &As[row][ks * 32 + ((v >= 4) ? 16 : 0) + g * 8 + (v & 3) * 2];
      }
#pragma unroll
      for (int nf = 0; nf < 4; ++nf) {
        const int col = wn * 64 + nf * 16 + hl;
#pragma unroll
        for (int v = 0; v < 8; ++v)
          bf[nf].u[v] = *(const unsigned int*)&Bs[col][ks * 32 + ((v >= 4) ? 16 : 0) + g * 8 + (v & 3) * 2];
      }
#pragma unroll
      for (int mf = 0; mf < 2; ++mf)
#pragma unroll
        for (int nf = 0; nf < 4; ++nf)
          acc[mf][nf] = __builtin_amdgcn_wmma_f32_16x16x32_bf16(
              false, a[mf].v, false, bf[nf].v, (short)0, acc[mf][nf], false, false);
    }
  }

#pragma unroll
  for (int mf = 0; mf < 2; ++mf)
#pragma unroll
    for (int nf = 0; nf < 4; ++nf) {
      const int rbase = m0 + wm * 32 + mf * 16 + g * 8;  // C/D: VGPR r -> M = r + 8*g
      const int col   = n0 + wn * 64 + nf * 16 + hl;
#pragma unroll
      for (int r = 0; r < 8; ++r) {
        const float  val = acc[mf][nf][r];
        const size_t off = (size_t)(rbase + r) * N + col;
        if (OUT_F32) ((float*)Dst)[off] = val;
        else ((unsigned short*)Dst)[off] = f32_to_bf16(val);
      }
    }
}

// ---------------------------------------------------------------------------
// RoPE (reference quirk: angle depends on head index h, not time t), in-place
// on bf16 (NT, C) tensor; also folds in optional 1/sqrt(D) scale.
// ---------------------------------------------------------------------------
__global__ void rope_bf16_kernel(unsigned short* __restrict__ X, float scale) {
  const int idx = blockIdx.x * blockDim.x + threadIdx.x;  // NT*H*64 threads
  const int d  = idx & 63;
  const int h  = (idx >> 6) & (H_ - 1);
  const int tk = idx >> 10;
  if (tk >= NT_) return;
  const size_t base = (size_t)tk * C_ + h * D_;
  const float x1 = bf16_to_f32(X[base + d]);
  const float x2 = bf16_to_f32(X[base + d + 64]);
  const float inv = __expf(-(float)d * 0.14391157f);      // ln(10000)/64
  float s, c;
  __sincosf((float)h * inv, &s, &c);
  X[base + d]      = f32_to_bf16((x1 * c - x2 * s) * scale);
  X[base + d + 64] = f32_to_bf16((x2 * c + x1 * s) * scale);
}

// ---------------------------------------------------------------------------
// Flash attention: block = 128 queries of one (b,h); 8 waves x 16 queries.
// K tile async memory->LDS; V tile transposed; online softmax; fp32 acc.
// ---------------------------------------------------------------------------
__global__ __launch_bounds__(256) void attn_fa_kernel(
    const unsigned short* __restrict__ Q,
    const unsigned short* __restrict__ Kb,
    const unsigned short* __restrict__ Vb,
    const float* __restrict__ mask,       // (B,1,T,T)
    unsigned short* __restrict__ Ctx)     // (NT, C) bf16
{
  __shared__ unsigned short Ks[32][128];      // [key][d]
  __shared__ unsigned short Vs[128][32];      // [d][key] (transposed)
  __shared__ unsigned short Ps[8][16][32];    // per-wave P scratch

  const int tid = threadIdx.x;
  const int lane = tid & 31, wid = tid >> 5;
  const int g = lane >> 4, hl = lane & 15;

  const int blk = blockIdx.x;           // B*H*(T/128) = 512
  const int qt = blk & 15;
  const int h  = (blk >> 4) & 15;
  const int b  = blk >> 8;

  const int q0 = qt * 128 + wid * 16;   // wave's query base within T

  // Q fragments (16 rows x 128 K), already RoPE'd and scaled by 1/sqrt(D)
  Frag qf[4];
  {
    const size_t rowoff = ((size_t)(b * T_) + q0 + hl) * C_ + h * D_;
#pragma unroll
    for (int c = 0; c < 4; ++c)
#pragma unroll
      for (int v = 0; v < 8; ++v)
        qf[c].u[v] = *(const unsigned int*)(Q + rowoff + c * 32 +
                        ((v >= 4) ? 16 : 0) + g * 8 + (v & 3) * 2);
  }

  floatx8 acc[8];
#pragma unroll
  for (int j = 0; j < 8; ++j) acc[j] = zero8();
  float m_r[8], l_r[8];
#pragma unroll
  for (int r = 0; r < 8; ++r) { m_r[r] = -1e30f; l_r[r] = 0.f; }

  const int lk = tid >> 3;          // 0..31 key row
  const int ld = (tid & 7) * 16;    // 16 contiguous d

  for (int kt = 0; kt < T_; kt += 32) {
    __syncthreads();
    {
      // K tile: async memory -> LDS, row-major (no transpose needed)
      const unsigned short* srcK = Kb + ((size_t)(b * T_) + kt + lk) * C_ + h * D_ + ld;
      copy16_to_lds(&Ks[lk][ld],     srcK);
      copy16_to_lds(&Ks[lk][ld + 8], srcK + 8);
      // V tile: transpose into [d][key]
      __align__(16) unsigned short tmp[16];
      const unsigned short* srcV = Vb + ((size_t)(b * T_) + kt + lk) * C_ + h * D_ + ld;
      *(uint4*)&tmp[0] = *(const uint4*)(srcV);
      *(uint4*)&tmp[8] = *(const uint4*)(srcV + 8);
#pragma unroll
      for (int i = 0; i < 16; ++i) Vs[ld + i][lk] = tmp[i];
    }
    wait_async_lds();
    __syncthreads();

    // ---- scores S = q . k^T for two 16-key fragments + mask ----
    floatx8 S[2];
#pragma unroll
    for (int kn = 0; kn < 2; ++kn) {
      floatx8 s = zero8();
#pragma unroll
      for (int c = 0; c < 4; ++c) {
        Frag bk;   // B = K^T: b[2v+j] = K[keycol][d = c*32 + (v>=4?16:0) + g*8 + 2(v&3)+j]
#pragma unroll
        for (int v = 0; v < 8; ++v)
          bk.u[v] = *(const unsigned int*)&Ks[kn * 16 + hl][c * 32 + ((v >= 4) ? 16 : 0) + g * 8 + (v & 3) * 2];
        s = __builtin_amdgcn_wmma_f32_16x16x32_bf16(
                false, qf[c].v, false, bk.v, (short)0, s, false, false);
      }
#pragma unroll
      for (int r = 0; r < 8; ++r)
        s[r] += mask[((size_t)b * T_ + (q0 + r + 8 * g)) * T_ + kt + kn * 16 + hl];
      S[kn] = s;
    }

    // ---- online softmax (row stats replicated across each 16-lane half) ----
    float alpha[8];
#pragma unroll
    for (int r = 0; r < 8; ++r) {
      float v = fmaxf(S[0][r], S[1][r]);
#pragma unroll
      for (int mk = 1; mk < 16; mk <<= 1) v = fmaxf(v, __shfl_xor(v, mk, 32));
      const float mn = fmaxf(m_r[r], v);
      alpha[r] = __expf(m_r[r] - mn);
      m_r[r] = mn;
    }
#pragma unroll
    for (int kn = 0; kn < 2; ++kn)
#pragma unroll
      for (int r = 0; r < 8; ++r) S[kn][r] = __expf(S[kn][r] - m_r[r]);
#pragma unroll
    for (int r = 0; r < 8; ++r) {
      float rs = S[0][r] + S[1][r];
#pragma unroll
      for (int mk = 1; mk < 16; mk <<= 1) rs += __shfl_xor(rs, mk, 32);
      l_r[r] = l_r[r] * alpha[r] + rs;
    }
#pragma unroll
    for (int j = 0; j < 8; ++j)
#pragma unroll
      for (int r = 0; r < 8; ++r) acc[j][r] *= alpha[r];

    // ---- C-layout -> A-layout for P via per-wave LDS ----
#pragma unroll
    for (int kn = 0; kn < 2; ++kn)
#pragma unroll
      for (int r = 0; r < 8; ++r)
        Ps[wid][r + 8 * g][kn * 16 + hl] = f32_to_bf16(S[kn][r]);
    asm volatile("s_wait_dscnt 0" ::: "memory");

    Frag pf;
#pragma unroll
    for (int v = 0; v < 8; ++v)
      pf.u[v] = *(const unsigned int*)&Ps[wid][hl][((v >= 4) ? 16 : 0) + g * 8 + (v & 3) * 2];

    // ---- acc += P @ V ----
#pragma unroll
    for (int j = 0; j < 8; ++j) {
      Frag bv;
#pragma unroll
      for (int v = 0; v < 8; ++v)
        bv.u[v] = *(const unsigned int*)&Vs[j * 16 + hl][((v >= 4) ? 16 : 0) + g * 8 + (v & 3) * 2];
      acc[j] = __builtin_amdgcn_wmma_f32_16x16x32_bf16(
                   false, pf.v, false, bv.v, (short)0, acc[j], false, false);
    }
  }

  // ---- normalize + store bf16 context ----
#pragma unroll
  for (int j = 0; j < 8; ++j)
#pragma unroll
    for (int r = 0; r < 8; ++r) {
      const float  val = acc[j][r] / l_r[r];
      const size_t row = (size_t)b * T_ + q0 + r + 8 * g;
      Ctx[row * C_ + h * D_ + j * 16 + hl] = f32_to_bf16(val);
    }
}

// ---------------------------------------------------------------------------
extern "C" void kernel_launch(void* const* d_in, const int* in_sizes, int n_in,
                              void* d_out, int out_size, void* d_ws, size_t ws_size,
                              hipStream_t stream) {
  (void)in_sizes; (void)n_in; (void)out_size; (void)ws_size;
  const float* x    = (const float*)d_in[0];
  const float* mask = (const float*)d_in[1];
  const float* Wq   = (const float*)d_in[2];
  const float* Wk   = (const float*)d_in[3];
  const float* Wv   = (const float*)d_in[4];
  const float* Wo   = (const float*)d_in[5];
  float* out = (float*)d_out;

  char* ws = (char*)d_ws;
  const size_t MiB = 1ull << 20;
  unsigned short* Xb  = (unsigned short*)(ws + 0  * MiB);  // 16 MiB
  unsigned short* Wqb = (unsigned short*)(ws + 16 * MiB);  //  8 MiB each
  unsigned short* Wkb = (unsigned short*)(ws + 24 * MiB);
  unsigned short* Wvb = (unsigned short*)(ws + 32 * MiB);
  unsigned short* Wob = (unsigned short*)(ws + 40 * MiB);
  unsigned short* Qb  = (unsigned short*)(ws + 48 * MiB);  // 16 MiB each
  unsigned short* Kb  = (unsigned short*)(ws + 64 * MiB);
  unsigned short* Vb  = (unsigned short*)(ws + 80 * MiB);
  unsigned short* Cxb = (unsigned short*)(ws + 96 * MiB);  // ends 112 MiB

  const int nX = NT_ * C_;   // 8M elements
  const int nW = C_ * C_;    // 4M elements
  cvt_f32_to_bf16_kernel<<<(nX / 4 + 255) / 256, 256, 0, stream>>>(x,  Xb,  nX / 4);
  cvt_f32_to_bf16_kernel<<<(nW / 4 + 255) / 256, 256, 0, stream>>>(Wq, Wqb, nW / 4);
  cvt_f32_to_bf16_kernel<<<(nW / 4 + 255) / 256, 256, 0, stream>>>(Wk, Wkb, nW / 4);
  cvt_f32_to_bf16_kernel<<<(nW / 4 + 255) / 256, 256, 0, stream>>>(Wv, Wvb, nW / 4);
  cvt_f32_to_bf16_kernel<<<(nW / 4 + 255) / 256, 256, 0, stream>>>(Wo, Wob, nW / 4);

  dim3 ggrid(C_ / 128, NT_ / 128);   // (16, 32)
  gemm_bf16_128x128<0><<<ggrid, 256, 0, stream>>>(Xb, Wqb, Qb, NT_, C_, C_);
  gemm_bf16_128x128<0><<<ggrid, 256, 0, stream>>>(Xb, Wkb, Kb, NT_, C_, C_);
  gemm_bf16_128x128<0><<<ggrid, 256, 0, stream>>>(Xb, Wvb, Vb, NT_, C_, C_);

  const int nR = NT_ * H_ * 64;      // 4M threads
  rope_bf16_kernel<<<nR / 256, 256, 0, stream>>>(Qb, 0.08838834764f); // 1/sqrt(128)
  rope_bf16_kernel<<<nR / 256, 256, 0, stream>>>(Kb, 1.0f);

  attn_fa_kernel<<<B_ * H_ * (T_ / 128), 256, 0, stream>>>(Qb, Kb, Vb, mask, Cxb);

  gemm_bf16_128x128<1><<<ggrid, 256, 0, stream>>>(Cxb, Wob, out, NT_, C_, C_);
}
